// GIN_6897717478006
// MI455X (gfx1250) — matrix operand order repaired and checked by reading, not compiled, not run on hardware
//
#include <hip/hip_runtime.h>
#include <hip/hip_bf16.h>

// ---------------------------------------------------------------------------
// GIN forward on gfx1250: bf16 WMMA GEMMs (32x64 per wave, 8 accumulators)
// + f32 atomics for graph scatter / pooling. M padded 50000 -> 50016.
// ---------------------------------------------------------------------------

typedef __attribute__((ext_vector_type(16))) __bf16 v16bf;
typedef __attribute__((ext_vector_type(8)))  float  v8f;

union Frag {
    unsigned int u[8];
    v16bf v;
};

#define ACT_NONE_F32  0
#define ACT_GELU_BF16 1
#define ACT_RELU_BF16 2

__device__ __forceinline__ float gelu_exact(float v) {
    return 0.5f * v * (1.0f + erff(v * 0.70710678118654752440f));
}

// ------------------------------ elementwise --------------------------------

__global__ void k_copy_f32(const float* __restrict__ in, float* __restrict__ out, size_t n) {
    size_t i = (size_t)blockIdx.x * blockDim.x + threadIdx.x;
    size_t stride = (size_t)gridDim.x * blockDim.x;
    for (; i < n; i += stride) out[i] = in[i];
}

__global__ void k_zero_f32(float* __restrict__ p, size_t n) {
    size_t i = (size_t)blockIdx.x * blockDim.x + threadIdx.x;
    size_t stride = (size_t)gridDim.x * blockDim.x;
    for (; i < n; i += stride) p[i] = 0.0f;
}

__global__ void k_zero_bf16(__hip_bfloat16* __restrict__ p, size_t n) {
    size_t i = (size_t)blockIdx.x * blockDim.x + threadIdx.x;
    size_t stride = (size_t)gridDim.x * blockDim.x;
    for (; i < n; i += stride) p[i] = __float2bfloat16(0.0f);
}

__global__ void k_cast_bf16(const float* __restrict__ in, __hip_bfloat16* __restrict__ out, size_t n) {
    size_t i = (size_t)blockIdx.x * blockDim.x + threadIdx.x;
    size_t stride = (size_t)gridDim.x * blockDim.x;
    for (; i < n; i += stride) out[i] = __float2bfloat16(in[i]);
}

// Wt[n*K + k] = bf16(W[k*N + n])  -- W is (K, N) row-major per reference.
__global__ void k_transpose_cast(const float* __restrict__ W, __hip_bfloat16* __restrict__ Wt,
                                 int K, int N) {
    size_t total = (size_t)K * N;
    size_t i = (size_t)blockIdx.x * blockDim.x + threadIdx.x;
    size_t stride = (size_t)gridDim.x * blockDim.x;
    for (; i < total; i += stride) {
        int k = (int)(i / N);
        int n = (int)(i % N);
        Wt[(size_t)n * K + k] = __float2bfloat16(W[i]);
    }
}

// ------------------------------ graph ops ----------------------------------

// agg[dst[e]] += h[src[e]]  (agg pre-initialized to h);  F multiple of 4.
__global__ void k_edge_scatter(const float* __restrict__ h, const int* __restrict__ src,
                               const int* __restrict__ dst, float* __restrict__ agg,
                               int E, int F) {
    int chunks = F >> 2;
    size_t total = (size_t)E * chunks;
    size_t i = (size_t)blockIdx.x * blockDim.x + threadIdx.x;
    size_t stride = (size_t)gridDim.x * blockDim.x;
    for (; i < total; i += stride) {
        int e = (int)(i / chunks);
        int j = (int)(i % chunks) << 2;
        const float4 v = *(const float4*)(h + (size_t)src[e] * F + j);
        float* p = agg + (size_t)dst[e] * F + j;
        atomicAdd(p + 0, v.x);
        atomicAdd(p + 1, v.y);
        atomicAdd(p + 2, v.z);
        atomicAdd(p + 3, v.w);
    }
}

// pool[batch[r]*768 + colOff + c] += h[r*F + c]
__global__ void k_pool(const float* __restrict__ h, const int* __restrict__ batch,
                       float* __restrict__ pool, int M, int F, int colOff) {
    size_t total = (size_t)M * F;
    size_t i = (size_t)blockIdx.x * blockDim.x + threadIdx.x;
    size_t stride = (size_t)gridDim.x * blockDim.x;
    for (; i < total; i += stride) {
        int r = (int)(i / F);
        int c = (int)(i % F);
        atomicAdd(&pool[(size_t)batch[r] * 768 + colOff + c], h[i]);
    }
}

// ------------------------------ batch norm ---------------------------------

__global__ void k_bn_stats(const float* __restrict__ X, float* __restrict__ mu,
                           float* __restrict__ rs, int M, int F) {
    __shared__ float sh[256];
    __shared__ float sh2[256];
    int c = blockIdx.x;
    float s = 0.0f, s2 = 0.0f;
    for (int r = threadIdx.x; r < M; r += blockDim.x) {
        float v = X[(size_t)r * F + c];
        s += v;
        s2 += v * v;
    }
    sh[threadIdx.x] = s;
    sh2[threadIdx.x] = s2;
    __syncthreads();
    for (int st = blockDim.x >> 1; st > 0; st >>= 1) {
        if ((int)threadIdx.x < st) {
            sh[threadIdx.x]  += sh[threadIdx.x + st];
            sh2[threadIdx.x] += sh2[threadIdx.x + st];
        }
        __syncthreads();
    }
    if (threadIdx.x == 0) {
        float m = sh[0] / (float)M;
        float var = sh2[0] / (float)M - m * m;
        mu[c] = m;
        rs[c] = rsqrtf(var + 1e-5f);
    }
}

// X = gelu(gamma*(X-mu)*rs + beta), in place.
__global__ void k_bn_gelu(float* __restrict__ X, const float* __restrict__ mu,
                          const float* __restrict__ rs, const float* __restrict__ g,
                          const float* __restrict__ b, size_t total, int F) {
    size_t i = (size_t)blockIdx.x * blockDim.x + threadIdx.x;
    size_t stride = (size_t)gridDim.x * blockDim.x;
    for (; i < total; i += stride) {
        int c = (int)(i % F);
        float v = g[c] * (X[i] - mu[c]) * rs[c] + b[c];
        X[i] = gelu_exact(v);
    }
}

// ------------------------------ WMMA GEMM ----------------------------------
// C[M x Nn] = A[M x K](bf16, row major) * Wt[Nn x K](bf16, = B^T) + bias.
// One wave owns a 32x64 output tile: 2 M-subtiles x 4 N-subtiles = 8
// accumulators; per K-step (32) it loads 2 A-frags + 4 B-frags (12 b128)
// and issues 8 v_wmma_f32_16x16x32_bf16 -> 1.5 loads/wmma.
// Requires M % 32 == 0, Nn % 64 == 0, K % 32 == 0.

__global__ void __launch_bounds__(128)
k_gemm_wmma(const __hip_bfloat16* __restrict__ A,
            const __hip_bfloat16* __restrict__ Wt,
            const float* __restrict__ bias,
            float* __restrict__ outF,
            __hip_bfloat16* __restrict__ outB,
            int M, int Nn, int K, int act) {
    const int lane = threadIdx.x & 31;
    const int wave = threadIdx.x >> 5;
    const int tiles64 = Nn >> 6;                 // 64-col strips
    const int numTiles = (M >> 5) * tiles64;     // 32 rows x 64 cols per wave
    const int tile = blockIdx.x * (blockDim.x >> 5) + wave;
    if (tile >= numTiles) return;                // wave-uniform: EXEC stays all-ones
    const int tm = tile / tiles64;
    const int tn = tile % tiles64;
    const int hi  = lane >> 4;                   // lane half (0: lanes 0-15, 1: 16-31)
    const int l16 = lane & 15;

    const size_t arow0 = (size_t)(tm * 32 + l16) * K;       // A rows for M-subtile 0/1
    const size_t arow1 = arow0 + (size_t)16 * K;
    size_t brow[4];
#pragma unroll
    for (int j = 0; j < 4; ++j)
        brow[j] = (size_t)(tn * 64 + j * 16 + l16) * K;     // Wt rows (= B columns)

    v8f acc[2][4];
#pragma unroll
    for (int i = 0; i < 2; ++i)
#pragma unroll
        for (int j = 0; j < 4; ++j)
            acc[i][j] = (v8f){};

    for (int kk = 0; kk < K; kk += 32) {
        // keep the A stream (HBM/L2) ahead of the XDL pipe
        if (kk + 64 < K) {
            __builtin_prefetch((const void*)(A + arow0 + kk + 64), 0, 3);
            __builtin_prefetch((const void*)(A + arow1 + kk + 64), 0, 3);
        }
        Frag a[2], b[4];
#pragma unroll
        for (int p = 0; p < 8; ++p) {
            // A 16x32 bf16 layout: vgpr p<4 -> K base 0, p>=4 -> K base 16;
            // lanes 16-31 offset K by +8 within each group; K pairs per dword.
            int ka = kk + ((p < 4) ? 0 : 16) + hi * 8 + ((p & 3) << 1);
            a[0].u[p] = *(const unsigned int*)(A + arow0 + ka);
            a[1].u[p] = *(const unsigned int*)(A + arow1 + ka);
            // B 32x16 bf16 layout: same K order as A (vgpr p<4 -> K base 0, p>=4 -> 16; lanes 16-31 +8).
            int kb = kk + ((p < 4) ? 0 : 16) + hi * 8 + ((p & 3) << 1);
#pragma unroll
            for (int j = 0; j < 4; ++j)
                b[j].u[p] = *(const unsigned int*)(Wt + brow[j] + kb);
        }
#pragma unroll
        for (int i = 0; i < 2; ++i)
#pragma unroll
            for (int j = 0; j < 4; ++j)
                acc[i][j] = __builtin_amdgcn_wmma_f32_16x16x32_bf16(
                    false, a[i].v, false, b[j].v, (short)0, acc[i][j], false, false);
    }

#pragma unroll
    for (int j = 0; j < 4; ++j) {
        const int col = tn * 64 + j * 16 + l16;
        const float bv = bias ? bias[col] : 0.0f;
#pragma unroll
        for (int i = 0; i < 2; ++i) {
#pragma unroll
            for (int r = 0; r < 8; ++r) {
                // D layout: VGPR r holds row (base + 8*hi + r), col = lane&15.
                int row = tm * 32 + i * 16 + hi * 8 + r;
                float v = acc[i][j][r] + bv;
                size_t idx = (size_t)row * Nn + col;
                if (act == ACT_GELU_BF16) {
                    outB[idx] = __float2bfloat16(gelu_exact(v));
                } else if (act == ACT_RELU_BF16) {
                    outB[idx] = __float2bfloat16(v > 0.0f ? v : 0.0f);
                } else {
                    outF[idx] = v;
                }
            }
        }
    }
}

// ------------------------------- host side ---------------------------------

extern "C" void kernel_launch(void* const* d_in, const int* in_sizes, int n_in,
                              void* d_out, int out_size, void* d_ws, size_t ws_size,
                              hipStream_t stream) {
    const int N = 50000, MP = 50016;   // MP = node count padded to multiple of 32
    const int H = 128, HL = 512, G = 512;
    const int E = in_sizes[1] / 2;

    const float* x     = (const float*)d_in[0];
    const int*   ei    = (const int*)d_in[1];
    const int*   srcI  = ei;
    const int*   dstI  = ei + E;
    const int*   batch = (const int*)d_in[2];
    const float* c1W1 = (const float*)d_in[3],  *c1b1 = (const float*)d_in[4];
    const float* c1W2 = (const float*)d_in[5],  *c1b2 = (const float*)d_in[6];
    const float* bn1g = (const float*)d_in[7],  *bn1b = (const float*)d_in[8];
    const float* c2W1 = (const float*)d_in[9],  *c2b1 = (const float*)d_in[10];
    const float* c2W2 = (const float*)d_in[11], *c2b2 = (const float*)d_in[12];
    const float* bn2g = (const float*)d_in[13], *bn2b = (const float*)d_in[14];
    const float* c3W1 = (const float*)d_in[15], *c3b1 = (const float*)d_in[16];
    const float* c3W2 = (const float*)d_in[17], *c3b2 = (const float*)d_in[18];
    const float* bn3g = (const float*)d_in[19], *bn3b = (const float*)d_in[20];
    const float* l1W  = (const float*)d_in[21], *l1b  = (const float*)d_in[22];
    const float* l2W  = (const float*)d_in[23], *l2b  = (const float*)d_in[24];

    // ---- workspace carve-out (~200 MB total) ----
    size_t off = 0;
    auto carve = [&](size_t bytes) -> void* {
        void* p = (char*)d_ws + off;
        off = (off + bytes + 255) & ~(size_t)255;
        return p;
    };
    float*          aggF  = (float*)         carve((size_t)N  * 128 * 4);
    __hip_bfloat16* aggB  = (__hip_bfloat16*)carve((size_t)MP * 128 * 2);
    __hip_bfloat16* gB    = (__hip_bfloat16*)carve((size_t)MP * 512 * 2);
    float*          oh    = (float*)         carve((size_t)MP * 512 * 4);
    float*          mu    = (float*)         carve(512 * 4);
    float*          rs    = (float*)         carve(512 * 4);
    float*          pool  = (float*)         carve((size_t)G * 768 * 4);
    __hip_bfloat16* poolB = (__hip_bfloat16*)carve((size_t)G * 768 * 2);
    __hip_bfloat16* l1o   = (__hip_bfloat16*)carve((size_t)G * 1024 * 2);
    __hip_bfloat16* tc1W1 = (__hip_bfloat16*)carve(128 * 128 * 2);
    __hip_bfloat16* tc1W2 = (__hip_bfloat16*)carve(128 * 128 * 2);
    __hip_bfloat16* tc2W1 = (__hip_bfloat16*)carve(128 * 128 * 2);
    __hip_bfloat16* tc2W2 = (__hip_bfloat16*)carve(128 * 128 * 2);
    __hip_bfloat16* tc3W1 = (__hip_bfloat16*)carve(128 * 512 * 2);
    __hip_bfloat16* tc3W2 = (__hip_bfloat16*)carve(512 * 512 * 2);
    __hip_bfloat16* tl1W  = (__hip_bfloat16*)carve(768 * 1024 * 2);
    __hip_bfloat16* tl2W  = (__hip_bfloat16*)carve(1024 * 512 * 2);
    (void)ws_size;

    auto blocksFor = [](size_t n, int t) { return (int)((n + t - 1) / t); };

    // ---- weight prep (transpose + bf16 cast) ----
    auto tw = [&](const float* W, __hip_bfloat16* Wt, int K, int Nn) {
        size_t tot = (size_t)K * Nn;
        k_transpose_cast<<<blocksFor(tot, 256), 256, 0, stream>>>(W, Wt, K, Nn);
    };
    tw(c1W1, tc1W1, 128, 128);  tw(c1W2, tc1W2, 128, 128);
    tw(c2W1, tc2W1, 128, 128);  tw(c2W2, tc2W2, 128, 128);
    tw(c3W1, tc3W1, 128, 512);  tw(c3W2, tc3W2, 512, 512);
    tw(l1W,  tl1W,  768, 1024); tw(l2W,  tl2W,  1024, 512);

    auto gemm = [&](const __hip_bfloat16* A, const __hip_bfloat16* Wt, const float* bias,
                    float* outF, __hip_bfloat16* outB, int M, int Nn, int K, int act) {
        int tiles = (M / 32) * (Nn / 64);
        int blocks = (tiles + 3) / 4;                 // 4 waves (32x64 tiles) per block
        k_gemm_wmma<<<blocks, 128, 0, stream>>>(A, Wt, bias, outF, outB, M, Nn, K, act);
    };

    // zero padded A rows (rows [N, MP) of aggB) and the pool accumulator
    k_zero_bf16<<<blocksFor((size_t)(MP - N) * 128, 256), 256, 0, stream>>>(
        aggB + (size_t)N * 128, (size_t)(MP - N) * 128);
    k_zero_f32<<<blocksFor((size_t)G * 768, 256), 256, 0, stream>>>(pool, (size_t)G * 768);

    const float* hin = x;   // current layer input (f32, [N x 128])
    const __hip_bfloat16* W1t[3] = { tc1W1, tc2W1, tc3W1 };
    const __hip_bfloat16* W2t[3] = { tc1W2, tc2W2, tc3W2 };
    const float* b1v[3] = { c1b1, c2b1, c3b1 };
    const float* b2v[3] = { c1b2, c2b2, c3b2 };
    const float* bng[3] = { bn1g, bn2g, bn3g };
    const float* bnb[3] = { bn1b, bn2b, bn3b };
    const int    Hhid[3]   = { H, H, HL };    // mlp hidden/out width per layer
    const int    colOff[3] = { 0, H, 2 * H };

    for (int layer = 0; layer < 3; ++layer) {
        const int Fh = Hhid[layer];
        // agg = hin + scatter_sum(hin[src] -> dst), f32 atomics (L2-resident)
        k_copy_f32<<<blocksFor((size_t)N * 128, 256), 256, 0, stream>>>(hin, aggF, (size_t)N * 128);
        k_edge_scatter<<<8192, 256, 0, stream>>>(hin, srcI, dstI, aggF, E, 128);
        k_cast_bf16<<<blocksFor((size_t)N * 128, 256), 256, 0, stream>>>(aggF, aggB, (size_t)N * 128);
        // hidden = gelu(agg @ W1 + b1)  -> bf16 (rows [N,MP) deterministic, unused)
        gemm(aggB, W1t[layer], b1v[layer], nullptr, gB, MP, Fh, 128, ACT_GELU_BF16);
        // conv out = hidden @ W2 + b2   -> f32 (BN needs exact column stats)
        gemm(gB, W2t[layer], b2v[layer], oh, nullptr, MP, Fh, Fh, ACT_NONE_F32);
        // batch norm + gelu over the N real rows, in place -> h
        k_bn_stats<<<Fh, 256, 0, stream>>>(oh, mu, rs, N, Fh);
        k_bn_gelu<<<blocksFor((size_t)N * Fh, 256), 256, 0, stream>>>(oh, mu, rs,
                bng[layer], bnb[layer], (size_t)N * Fh, Fh);
        // global add pool into concat columns [p1 | p2 | p3]
        k_pool<<<blocksFor((size_t)N * Fh, 256), 256, 0, stream>>>(oh, batch, pool, N, Fh, colOff[layer]);
        hin = oh;   // next layer aggregates h (layers 1,2 have Fh==128)
    }

    // ----------------------------- classifier ------------------------------
    k_cast_bf16<<<blocksFor((size_t)G * 768, 256), 256, 0, stream>>>(pool, poolB, (size_t)G * 768);
    gemm(poolB, tl1W, l1b, nullptr, l1o, G, 1024, 768, ACT_RELU_BF16);          // relu(p @ lin1W + b)
    gemm(l1o,   tl2W, l2b, (float*)d_out, nullptr, G, 512, 1024, ACT_NONE_F32); // logits -> d_out
}
